// MPNNEncoder_75393855914544
// MI455X (gfx1250) — matrix operand order repaired and checked by reading, not compiled, not run on hardware
//
#include <hip/hip_runtime.h>

typedef __attribute__((ext_vector_type(16))) __bf16    v16bf;
typedef __attribute__((ext_vector_type(8)))  float     v8f;
typedef __attribute__((ext_vector_type(4)))  unsigned  u32x4;
typedef __attribute__((ext_vector_type(4)))  int       v4i;
typedef __attribute__((ext_vector_type(2)))  float     f2;
typedef __attribute__((ext_vector_type(4)))  float     f4;
typedef unsigned short u16;

#define NN 200000
#define EEDG 800000
#define GG 8192
#define HH 256

// ---- async global->LDS path (guarded: falls back to scalar staging) ----
#if defined(__has_builtin)
#  if __has_builtin(__builtin_amdgcn_global_load_async_to_lds_b128) && \
      __has_builtin(__builtin_amdgcn_s_wait_asynccnt)
#    define USE_ASYNC 1
#  endif
#endif
#ifndef USE_ASYNC
#  define USE_ASYNC 0
#endif

#if USE_ASYNC
typedef __attribute__((address_space(1))) v4i* gv4p;   // global int4*
typedef __attribute__((address_space(3))) v4i* lv4p;   // LDS int4*
__device__ __forceinline__ void async_cp16(const void* g, void* l) {
  __builtin_amdgcn_global_load_async_to_lds_b128((gv4p)g, (lv4p)l, 0, 0);
}
#endif

__device__ __forceinline__ u16 bfbits(float f) {
  __bf16 b = (__bf16)f;               // native v_cvt on gfx1250, RNE
  u16 u; __builtin_memcpy(&u, &b, 2); return u;
}
__device__ __forceinline__ float bf2f(unsigned bits16) {
  union { unsigned u; float f; } c; c.u = bits16 << 16; return c.f;
}

// B-tile LDS index: natural [k][n], stride 136 floats + 4-float swizzle per K-half
__device__ __forceinline__ int bidx(int k, int n) {
  return k * 136 + ((k >> 4) & 1) * 4 + n;
}

// ---------------------------------------------------------------------------
// Tiled GEMM: C[M,256] = epilogue( prologueA[M,K] @ W[K,256] + bias )
//   AMODE 0: A = Aptr (f32, lda=K)     AMODE 1: A = (1+eps[layer])*Aptr + AGG
//   EMODE 0: C = acc+bias ; 1: relu ; 2: RES + relu(acc+bias)
//   OUTBF 1: store bf16 bits to Cb, else f32 to Cf
// Block: 256 threads (8 waves); tile 64(M) x 128(N); BK=32; 4 WMMA/wave/ktile.
// ---------------------------------------------------------------------------
template<int AMODE, int EMODE, int OUTBF>
__global__ __launch_bounds__(256)
void gemm256_k(const float* __restrict__ A, const float* __restrict__ AGG,
               const float* __restrict__ EPS, int layer,
               const float* __restrict__ W, const float* __restrict__ bias,
               const float* __restrict__ RES,
               float* __restrict__ Cf, u16* __restrict__ Cb,
               int M, int K)
{
  __shared__ float As[64 * 36];        // [row][k], stride 36 (144B, 16B-aligned)
  __shared__ float Bs[32 * 136 + 8];   // [k][n], stride 136 + half-swizzle

  const int tid  = threadIdx.x;
  const int row0 = blockIdx.x * 64;
  const int col0 = blockIdx.y * 128;
  const float scale = (AMODE == 1) ? (1.0f + EPS[layer]) : 0.0f;

  const int lane = tid & 31;
  const int wave = tid >> 5;
  const int half = lane >> 4;
  const int r16  = lane & 15;
  const int wm   = wave >> 2;
  const int wn   = wave & 3;

  v8f acc[2][2] = {};

  for (int k0 = 0; k0 < K; k0 += 32) {
    const bool fullK = (k0 + 32 <= K);

    // ---- stage A tile 64x32 (f32) ----
#if USE_ASYNC
    if (AMODE == 0 && fullK) {
      int rr = tid >> 2, kk = (tid & 3) * 8;
      const float* g = A + (size_t)(row0 + rr) * K + (k0 + kk);
      float*       l = &As[rr * 36 + kk];
      async_cp16(g, l);
      async_cp16(g + 4, l + 4);
    } else
#endif
    {
      for (int e = tid; e < 64 * 32; e += 256) {
        int rr = e >> 5, kk = e & 31, kg = k0 + kk;
        float v = 0.0f;
        if (AMODE == 0) { if (kg < K) v = A[(size_t)(row0 + rr) * K + kg]; }
        else            { size_t ix = (size_t)(row0 + rr) * HH + kg;
                          v = scale * A[ix] + AGG[ix]; }
        As[rr * 36 + kk] = v;
      }
    }

    // ---- stage B tile 32x128 (f32, natural layout) ----
#if USE_ASYNC
    if (fullK) {
#pragma unroll
      for (int it = 0; it < 2; it++) {
        int e = tid * 4 + it * 1024;
        int k = e >> 7, n = e & 127;
        async_cp16(W + (size_t)(k0 + k) * HH + col0 + n, &Bs[bidx(k, n)]);
      }
    } else
#endif
    {
      for (int e = tid; e < 32 * 128; e += 256) {
        int n = e & 127, k = e >> 7, kg = k0 + k;
        Bs[bidx(k, n)] = (kg < K) ? W[(size_t)kg * HH + col0 + n] : 0.0f;
      }
    }

#if USE_ASYNC
    __builtin_amdgcn_s_wait_asynccnt(0);
#endif
    __syncthreads();

    // ---- build bf16 fragments (contiguous k = 16*half + s, same in A and B), native cvt ----
    v16bf afr[2], bfr[2];
#pragma unroll
    for (int mi = 0; mi < 2; mi++) {
      int arow = wm * 32 + mi * 16 + r16;
#pragma unroll
      for (int vv = 0; vv < 8; vv++) {
        int ko = 16 * half + 2 * vv;
        f2 x = *(const f2*)&As[arow * 36 + ko];
        afr[mi][2 * vv]     = (__bf16)x.x;
        afr[mi][2 * vv + 1] = (__bf16)x.y;
      }
    }
#pragma unroll
    for (int ni = 0; ni < 2; ni++) {
      int bcol = wn * 32 + ni * 16 + r16;
#pragma unroll
      for (int vv = 0; vv < 8; vv++) {
        int ko = 16 * half + 2 * vv;
        int bi = bidx(ko, bcol);
        bfr[ni][2 * vv]     = (__bf16)Bs[bi];
        bfr[ni][2 * vv + 1] = (__bf16)Bs[bi + 136];
      }
    }

#pragma unroll
    for (int mi = 0; mi < 2; mi++)
#pragma unroll
      for (int ni = 0; ni < 2; ni++)
        acc[mi][ni] = __builtin_amdgcn_wmma_f32_16x16x32_bf16(
            false, afr[mi], false, bfr[ni], (short)0, acc[mi][ni], false, false);

    __syncthreads();
  }

  // ---- epilogue: C VGPR i -> row = i + 8*half, col = r16 ----
#pragma unroll
  for (int mi = 0; mi < 2; mi++) {
#pragma unroll
    for (int ni = 0; ni < 2; ni++) {
      int crow0 = row0 + wm * 32 + mi * 16;
      int ccol  = col0 + wn * 32 + ni * 16 + r16;
      float bb = bias[ccol];
#pragma unroll
      for (int i = 0; i < 8; i++) {
        int rrow = crow0 + i + 8 * half;
        float v = acc[mi][ni][i] + bb;
        if (EMODE >= 1) v = fmaxf(v, 0.0f);
        size_t oi = (size_t)rrow * HH + ccol;
        if (EMODE == 2) v += RES[oi];
        if (OUTBF) Cb[oi] = bfbits(v);
        else       Cf[oi] = v;
      }
    }
  }
}

// ---------------------------------------------------------------------------
// msg = relu(h[src] + e); agg[dst] += msg   (one thread = 8 features of 1 edge)
// ---------------------------------------------------------------------------
__global__ __launch_bounds__(256)
void gather_scatter_k(const float* __restrict__ h, const u16* __restrict__ e,
                      const int* __restrict__ src, const int* __restrict__ dst,
                      float* __restrict__ agg)
{
  size_t idx  = (size_t)blockIdx.x * blockDim.x + threadIdx.x;  // EEDG*32 threads
  size_t edge = idx >> 5;
  int    c8   = (int)(idx & 31) * 8;
  int s = src[edge], d = dst[edge];

  u32x4 epk = *(const u32x4*)(e + edge * HH + c8);              // 8 bf16
  const float* hp = h + (size_t)s * HH + c8;
  f4 h0 = *(const f4*)hp;
  f4 h1 = *(const f4*)(hp + 4);
  float hv[8] = { h0.x, h0.y, h0.z, h0.w, h1.x, h1.y, h1.z, h1.w };

  float* ap = agg + (size_t)d * HH + c8;
#pragma unroll
  for (int j = 0; j < 4; j++) {
    float e0 = bf2f(epk[j] & 0xFFFFu);
    float e1 = bf2f(epk[j] >> 16);
    float m0 = fmaxf(hv[2 * j]     + e0, 0.0f);
    float m1 = fmaxf(hv[2 * j + 1] + e1, 0.0f);
    __hip_atomic_fetch_add(&ap[2 * j],     m0, __ATOMIC_RELAXED, __HIP_MEMORY_SCOPE_AGENT);
    __hip_atomic_fetch_add(&ap[2 * j + 1], m1, __ATOMIC_RELAXED, __HIP_MEMORY_SCOPE_AGENT);
  }
}

__global__ __launch_bounds__(256)
void pool_sum_k(const float* __restrict__ h, const int* __restrict__ batch,
                float* __restrict__ sums)
{
  size_t idx  = (size_t)blockIdx.x * blockDim.x + threadIdx.x;  // NN*32 threads
  size_t node = idx >> 5;
  int    c8   = (int)(idx & 31) * 8;
  int b = batch[node];
  const float* hp = h + node * HH + c8;
  float* sp = sums + (size_t)b * HH + c8;
#pragma unroll
  for (int j = 0; j < 8; j++)
    __hip_atomic_fetch_add(&sp[j], hp[j], __ATOMIC_RELAXED, __HIP_MEMORY_SCOPE_AGENT);
}

__global__ __launch_bounds__(256)
void pool_cnt_k(const int* __restrict__ batch, float* __restrict__ cnt, int n)
{
  int i = blockIdx.x * blockDim.x + threadIdx.x;
  if (i < n)
    __hip_atomic_fetch_add(&cnt[batch[i]], 1.0f, __ATOMIC_RELAXED, __HIP_MEMORY_SCOPE_AGENT);
}

__global__ __launch_bounds__(256)
void pool_div_k(float* __restrict__ sums, const float* __restrict__ cnt)
{
  size_t i = (size_t)blockIdx.x * blockDim.x + threadIdx.x;    // GG*HH threads
  sums[i] = sums[i] / fmaxf(cnt[i >> 8], 1.0f);
}

// in-place LayerNorm over rows of 256; one wave32 per row
__global__ __launch_bounds__(256)
void layernorm_k(float* __restrict__ o, const float* __restrict__ g,
                 const float* __restrict__ b)
{
  int wave = threadIdx.x >> 5, lane = threadIdx.x & 31;
  int row  = blockIdx.x * 8 + wave;
  float* rp = o + (size_t)row * HH;
  float vals[8], s = 0.0f, s2 = 0.0f;
#pragma unroll
  for (int j = 0; j < 8; j++) {
    float v = rp[lane + 32 * j];
    vals[j] = v; s += v; s2 += v * v;
  }
#pragma unroll
  for (int m = 16; m > 0; m >>= 1) {
    s  += __shfl_xor(s,  m, 32);
    s2 += __shfl_xor(s2, m, 32);
  }
  float mu  = s * (1.0f / HH);
  float var = s2 * (1.0f / HH) - mu * mu;
  float inv = rsqrtf(var + 1e-5f);
#pragma unroll
  for (int j = 0; j < 8; j++) {
    int c = lane + 32 * j;
    rp[c] = (vals[j] - mu) * inv * g[c] + b[c];
  }
}

extern "C" void kernel_launch(void* const* d_in, const int* in_sizes, int n_in,
                              void* d_out, int out_size, void* d_ws, size_t ws_size,
                              hipStream_t stream) {
  const float* x   = (const float*)d_in[0];
  const float* ea  = (const float*)d_in[1];
  const int*   ei  = (const int*)d_in[2];   // [2,E]: src then dst
  const int*   bat = (const int*)d_in[3];
  const float* aw  = (const float*)d_in[4];
  const float* ab  = (const float*)d_in[5];
  const float* ew  = (const float*)d_in[6];
  const float* eb  = (const float*)d_in[7];
  const float* eps = (const float*)d_in[8];
  const float* cw1 = (const float*)d_in[9];
  const float* cb1 = (const float*)d_in[10];
  const float* cw2 = (const float*)d_in[11];
  const float* cb2 = (const float*)d_in[12];
  const float* ow1 = (const float*)d_in[13];
  const float* ob1 = (const float*)d_in[14];
  const float* ow2 = (const float*)d_in[15];
  const float* ob2 = (const float*)d_in[16];
  const float* lng = (const float*)d_in[17];
  const float* lnb = (const float*)d_in[18];
  (void)in_sizes; (void)n_in; (void)out_size; (void)ws_size;

  char* ws = (char*)d_ws;
  float* h    = (float*)ws;  ws += (size_t)NN * HH * 4;
  float* agg  = (float*)ws;  ws += (size_t)NN * HH * 4;
  float* tbuf = (float*)ws;  ws += (size_t)NN * HH * 4;
  u16*   ebuf = (u16*)ws;    ws += (size_t)EEDG * HH * 2;
  float* sums = (float*)ws;  ws += (size_t)GG * HH * 4;
  float* cnt  = (float*)ws;  ws += (size_t)GG * 4;
  float* t2   = (float*)ws;  ws += (size_t)GG * HH * 4;
  float* out  = (float*)d_out;

  const dim3 blk(256);
  const dim3 gN(NN / 64, 2), gE(EEDG / 64, 2), gG(GG / 64, 2);

  // encoders
  gemm256_k<0, 0, 0><<<gN, blk, 0, stream>>>(x,  nullptr, nullptr, 0, aw, ab,
                                             nullptr, h, nullptr, NN, 84);
  gemm256_k<0, 0, 1><<<gE, blk, 0, stream>>>(ea, nullptr, nullptr, 0, ew, eb,
                                             nullptr, nullptr, ebuf, EEDG, 14);

  // 3x GINEConv
  for (int l = 0; l < 3; l++) {
    (void)hipMemsetAsync(agg, 0, (size_t)NN * HH * 4, stream);
    gather_scatter_k<<<(EEDG * 32) / 256, blk, 0, stream>>>(h, ebuf, ei, ei + EEDG, agg);
    gemm256_k<1, 1, 0><<<gN, blk, 0, stream>>>(h, agg, eps, l,
                                               cw1 + (size_t)l * HH * HH, cb1 + (size_t)l * HH,
                                               nullptr, tbuf, nullptr, NN, HH);
    gemm256_k<0, 2, 0><<<gN, blk, 0, stream>>>(tbuf, nullptr, nullptr, 0,
                                               cw2 + (size_t)l * HH * HH, cb2 + (size_t)l * HH,
                                               h, h, nullptr, NN, HH);
  }

  // global mean pool
  (void)hipMemsetAsync(sums, 0, (size_t)(GG * HH + GG) * 4, stream);
  pool_sum_k<<<(NN * 32) / 256, blk, 0, stream>>>(h, bat, sums);
  pool_cnt_k<<<(NN + 255) / 256, blk, 0, stream>>>(bat, cnt, NN);
  pool_div_k<<<(GG * HH) / 256, blk, 0, stream>>>(sums, cnt);

  // output MLP + LayerNorm
  gemm256_k<0, 1, 0><<<gG, blk, 0, stream>>>(sums, nullptr, nullptr, 0, ow1, ob1,
                                             nullptr, t2, nullptr, GG, HH);
  gemm256_k<0, 0, 0><<<gG, blk, 0, stream>>>(t2, nullptr, nullptr, 0, ow2, ob2,
                                             nullptr, out, nullptr, GG, HH);
  layernorm_k<<<GG / 8, blk, 0, stream>>>(out, lng, lnb);
}
